// SelfAttention_86028194939077
// MI455X (gfx1250) — matrix operand order repaired and checked by reading, not compiled, not run on hardware
//
#include <hip/hip_runtime.h>

// Problem constants (N,S,E,H,D) from the reference
#define NB 4
#define SS 2048
#define EE 1024
#define HH 16
#define DDIM 64

typedef __bf16 bf16_t;
typedef __attribute__((ext_vector_type(16))) __bf16 v16bf;
typedef __attribute__((ext_vector_type(8)))  __bf16 v8bf;
typedef __attribute__((ext_vector_type(4)))  __bf16 v4bf;
typedef __attribute__((ext_vector_type(8)))  float  v8f;
typedef __attribute__((ext_vector_type(4)))  float  v4f;
typedef __attribute__((ext_vector_type(4)))  unsigned int u32x4;
typedef __attribute__((ext_vector_type(8)))  int i32x8;
typedef __attribute__((ext_vector_type(4)))  int i32x4;

union Frag16 { v16bf v; v8bf h[2]; };

__device__ __forceinline__ v8f wmma_bf16(const Frag16& a, const Frag16& b, v8f c) {
  return __builtin_amdgcn_wmma_f32_16x16x32_bf16(false, a.v, false, b.v, (short)0, c, false, false);
}

// ---------------------------------------------------------------------------
// TDM: stage a 2D tile (tile_x contiguous elems per row, tile_y rows, row
// pitch stride_elems, bf16 data) from global into LDS, with an LDS pad of
// pad_amount(+1) DWORDs inserted every (1<<pad_interval)*2 DWORDs (one row).
// D# layout per CDNA5 ISA. Issued once per wave; caller
// gates to one wave per block and synchronizes via TENSORcnt + barriers.
// ---------------------------------------------------------------------------
__device__ __forceinline__ void tdm_stage_2d(const bf16_t* gsrc, const bf16_t* ldst,
                                             unsigned tile_x, unsigned tile_y,
                                             unsigned long long stride_elems,
                                             unsigned pad_interval, unsigned pad_amount) {
  const unsigned long long ga = (unsigned long long)(size_t)gsrc;
  const unsigned lds_addr = (unsigned)(size_t)ldst;   // low 32 bits = LDS offset
  u32x4 g0;
  g0.x = 0x1u;                                        // count=1, user descriptor
  g0.y = lds_addr;                                    // [63:32] lds_addr
  g0.z = (unsigned)ga;                                // [95:64] global_addr lo
  g0.w = (unsigned)(ga >> 32) | (2u << 30);           // [120:96] ga hi, type=2

  i32x8 g1;
  // [15:0] wg_mask=0 | [17:16] data_size=1(2B) | [20] pad_en | [24:22] pi | [31:25] pa
  g1[0] = (int)((1u << 16) | (1u << 20) | (pad_interval << 22) | (pad_amount << 25));
  g1[1] = (int)((tile_x & 0xffffu) << 16);            // abar=0, tensor_dim0[15:0]
  g1[2] = (int)(((tile_x >> 16) & 0xffffu) | ((tile_y & 0xffffu) << 16)); // td0 hi, td1 lo
  g1[3] = (int)(((tile_y >> 16) & 0xffffu) | ((tile_x & 0xffffu) << 16)); // td1 hi, tile_dim0
  g1[4] = (int)(tile_y & 0xffffu);                    // tile_dim1 ; tile_dim2=0
  g1[5] = (int)(unsigned)(stride_elems & 0xffffffffull);        // dim0_stride[31:0]
  g1[6] = (int)(unsigned)((stride_elems >> 32) & 0xffffull);    // dim0_stride[47:32]
  g1[7] = 0;

  const i32x4 z4 = {0, 0, 0, 0};
#if defined(__clang_major__) && (__clang_major__ >= 23)
  const i32x8 z8 = {0, 0, 0, 0, 0, 0, 0, 0};
  __builtin_amdgcn_tensor_load_to_lds(g0, g1, z4, z4, z8, 0);
#else
  __builtin_amdgcn_tensor_load_to_lds(g0, g1, z4, z4, 0);
#endif
}

// ---------------------------------------------------------------------------
// fp32 -> bf16 conversion, 4 elements per thread
// ---------------------------------------------------------------------------
__global__ __launch_bounds__(256) void f32_to_bf16(const float* __restrict__ src,
                                                   bf16_t* __restrict__ dst, int n4) {
  int i = blockIdx.x * blockDim.x + threadIdx.x;
  if (i >= n4) return;
  v4f v = ((const v4f*)src)[i];
  v4bf o;
  o.x = (bf16_t)v.x; o.y = (bf16_t)v.y; o.z = (bf16_t)v.z; o.w = (bf16_t)v.w;
  ((v4bf*)dst)[i] = o;
}

// ---------------------------------------------------------------------------
// GEMM:  out[m, e] = sum_k X[m,k] * W[e,k] + bias[e]
// One wave computes a 16x64 output tile; all 8 waves of a block share the same
// 64-column W slice, so W k-slabs (64 rows x 32 k) are staged block-wide into
// LDS by the TDM (double-buffered), and B fragments come from ds_load_b128.
// MODE 0: store bf16 row-major   MODE 1: store bf16 as VT[n,h,d,s]
// MODE 2: store fp32 row-major
// ---------------------------------------------------------------------------
template <int MODE>
__global__ __launch_bounds__(256) void gemm16x64(const bf16_t* __restrict__ X,
                                                 const bf16_t* __restrict__ W,
                                                 const float* __restrict__ bias,
                                                 void* __restrict__ out,
                                                 int M, int K, int Nn) {
  __shared__ __align__(16) bf16_t wbuf[2][64 * 40];   // 80B row pitch (32 data + 8 pad halfs)
  const int lane = threadIdx.x & 31;
  const int w    = threadIdx.x >> 5;
  const int wid  = (blockIdx.x * blockDim.x + threadIdx.x) >> 5;
  const int mTiles = M >> 4;
  const int tm = wid % mTiles;
  const int tn = wid / mTiles;                        // same for all 8 waves in block
  const int m0 = tm << 4;
  const int n0 = tn << 6;
  const int cl  = lane & 15;
  const int hiB = lane >> 4;
  const int aoff = hiB ? 8 : 0;
  const int koff = hiB ? 16 : 0;
  const int row  = m0 + cl;

  // prologue: stage first W slab
  if (w == 0) {
    tdm_stage_2d(W + (size_t)n0 * K, &wbuf[0][0], 32u, 64u,
                 (unsigned long long)K, 3u, 3u);      // rows: 16 DW + 4 DW pad
    __builtin_amdgcn_s_wait_tensorcnt(0);
  }
  __syncthreads();

  v8f acc[4] = {};
  int buf = 0;
  for (int kk = 0; kk < K; kk += 32) {
    if (w == 0 && kk + 32 < K)
      tdm_stage_2d(W + (size_t)n0 * K + kk + 32, &wbuf[buf ^ 1][0], 32u, 64u,
                   (unsigned long long)K, 3u, 3u);

    Frag16 a;
    const bf16_t* ap = X + (size_t)row * K + kk;
    a.h[0] = *(const v8bf*)(ap + aoff);
    a.h[1] = *(const v8bf*)(ap + 16 + aoff);

    const bf16_t* wl = &wbuf[buf][0];
#pragma unroll
    for (int t = 0; t < 4; ++t) {
      const bf16_t* bp = wl + (t * 16 + cl) * 40 + aoff;
      Frag16 b;
      b.h[0] = *(const v8bf*)(bp);
      b.h[1] = *(const v8bf*)(bp + 16);
      acc[t] = wmma_bf16(a, b, acc[t]);
    }

    __syncthreads();                                  // all waves done with wbuf[buf]
    if (w == 0 && kk + 32 < K) __builtin_amdgcn_s_wait_tensorcnt(0);
    __syncthreads();                                  // next slab visible
    buf ^= 1;
  }

  const int rbase = m0 + (hiB ? 8 : 0);
#pragma unroll
  for (int t = 0; t < 4; ++t) {
    const int col = n0 + t * 16 + cl;
    const float bb = bias[col];
#pragma unroll
    for (int r = 0; r < 8; ++r) {
      const float val = acc[t][r] + bb;
      const int rr = rbase + r;
      if (MODE == 2) {
        ((float*)out)[(size_t)rr * Nn + col] = val;
      } else if (MODE == 0) {
        ((bf16_t*)out)[(size_t)rr * Nn + col] = (bf16_t)val;
      } else {
        const int nn = rr >> 11;          // / S
        const int s  = rr & (SS - 1);
        const int h  = col >> 6;
        const int d  = col & (DDIM - 1);
        ((bf16_t*)out)[((((size_t)nn * HH + h) * DDIM) + d) * SS + s] = (bf16_t)val;
      }
    }
  }
}

// ---------------------------------------------------------------------------
// Flash-style attention. One wave per (n, h, 16-query tile); the 8 waves of a
// block are 8 consecutive q-tiles of the SAME (n,h), so the 32-key K and V
// slabs are TDM-staged into LDS once per block (double-buffered).
// scores: S = Q(16xD) @ K^T ; online softmax in log2 domain
// (faithful quirk: masked scores set to 1e-20 BEFORE the /sqrt(E)=32 scale).
// O += P(16x32) @ V(32x16-d-tile); P goes C-layout -> A-layout via LDS.
// ---------------------------------------------------------------------------
__global__ __launch_bounds__(256) void attn_fwd(const bf16_t* __restrict__ Q,
                                                const bf16_t* __restrict__ Kp,
                                                const bf16_t* __restrict__ VT,
                                                const int* __restrict__ mask,
                                                bf16_t* __restrict__ ctx) {
  __shared__ __align__(16) bf16_t pshare[8][16 * 48];   // per-wave P tile, 96B pitch
  __shared__ __align__(16) bf16_t kbuf[2][32 * 72];     // K slab: 32 keys x 64 d, 144B pitch
  __shared__ __align__(16) bf16_t vbuf[2][64 * 40];     // V slab: 64 d x 32 keys, 80B pitch
  const int lane = threadIdx.x & 31;
  const int w    = threadIdx.x >> 5;
  const int wid  = (blockIdx.x * blockDim.x + threadIdx.x) >> 5;
  const int qt = wid & 127;            // S/16 tiles (8 consecutive per block)
  const int h  = (wid >> 7) & 15;
  const int n  = wid >> 11;
  const int q0 = qt << 4;
  const int cl  = lane & 15;
  const int hiB = lane >> 4;
  const int aoff = hiB ? 8 : 0;
  const int koff = hiB ? 16 : 0;
  const float SCL = 0.04508422f;       // log2(e) / sqrt(E) = log2(e)/32
  const float MASKED = 1e-20f * 0.04508422f;

  const bf16_t* kg = Kp + (size_t)n * SS * EE + h * DDIM;              // + kb*EE
  const bf16_t* vg = VT + (((size_t)n * HH + h) * DDIM) * SS;          // + kb

  // prologue: stage key-block 0
  if (w == 0) {
    tdm_stage_2d(kg, &kbuf[0][0], 64u, 32u, (unsigned long long)EE, 4u, 3u);
    tdm_stage_2d(vg, &vbuf[0][0], 32u, 64u, (unsigned long long)SS, 3u, 3u);
    __builtin_amdgcn_s_wait_tensorcnt(0);
  }

  // Q fragments for d-slices [0,32) and [32,64)
  Frag16 qa[2];
  {
    const bf16_t* qp = Q + ((size_t)n * SS + q0 + cl) * EE + h * DDIM;
#pragma unroll
    for (int dd = 0; dd < 2; ++dd) {
      qa[dd].h[0] = *(const v8bf*)(qp + dd * 32 + aoff);
      qa[dd].h[1] = *(const v8bf*)(qp + dd * 32 + 16 + aoff);
    }
  }

  v8f o[4] = {};
  float mrun[8], lrun[8];
#pragma unroll
  for (int r = 0; r < 8; ++r) { mrun[r] = -3.0e38f; lrun[r] = 0.0f; }

  const int* mrowbase = mask + ((size_t)n * SS + q0 + (hiB ? 8 : 0)) * SS;
  bf16_t* mysh = &pshare[w][0];

  __syncthreads();
  int buf = 0;
  for (int kb = 0; kb < SS; kb += 32) {
    // async-stage the next key-block while computing this one
    if (w == 0 && kb + 32 < SS) {
      tdm_stage_2d(kg + (size_t)(kb + 32) * EE, &kbuf[buf ^ 1][0], 64u, 32u,
                   (unsigned long long)EE, 4u, 3u);
      tdm_stage_2d(vg + (kb + 32), &vbuf[buf ^ 1][0], 32u, 64u,
                   (unsigned long long)SS, 3u, 3u);
    }
    const bf16_t* kl = &kbuf[buf][0];
    const bf16_t* vl = &vbuf[buf][0];

    // ---- scores: two 16x16 key sub-tiles, K-dim = D via 2 WMMA steps ----
    v8f s0 = {}, s1 = {};
#pragma unroll
    for (int dd = 0; dd < 2; ++dd) {
      {
        const bf16_t* bp = kl + cl * 72 + dd * 32 + aoff;
        Frag16 b;
        b.h[0] = *(const v8bf*)bp;
        b.h[1] = *(const v8bf*)(bp + 16);
        s0 = wmma_bf16(qa[dd], b, s0);
      }
      {
        const bf16_t* bp = kl + (16 + cl) * 72 + dd * 32 + aoff;
        Frag16 b;
        b.h[0] = *(const v8bf*)bp;
        b.h[1] = *(const v8bf*)(bp + 16);
        s1 = wmma_bf16(qa[dd], b, s1);
      }
    }

    // ---- mask + scale into log2 domain ----
    if (kb + 32 < SS)
      __builtin_prefetch(mrowbase + kb + 32 + cl, 0, 3);   // global_prefetch_b8
    float t0a[8], t1a[8], bm[8];
#pragma unroll
    for (int r = 0; r < 8; ++r) {
      const int* mp = mrowbase + (size_t)r * SS + kb + cl;
      t0a[r] = (mp[0]  != 0) ? s0[r] * SCL : MASKED;
      t1a[r] = (mp[16] != 0) ? s1[r] * SCL : MASKED;
      bm[r]  = fmaxf(t0a[r], t1a[r]);
    }
#pragma unroll
    for (int off = 1; off < 16; off <<= 1)
#pragma unroll
      for (int r = 0; r < 8; ++r)
        bm[r] = fmaxf(bm[r], __shfl_xor(bm[r], off, 32));

    // ---- online softmax update; write P (bf16) into LDS C-layout ----
    float rs[8];
#pragma unroll
    for (int r = 0; r < 8; ++r) {
      const float Mn    = fmaxf(mrun[r], bm[r]);
      const float alpha = exp2f(mrun[r] - Mn);
      mrun[r] = Mn;
      lrun[r] *= alpha;
#pragma unroll
      for (int t = 0; t < 4; ++t) o[t][r] *= alpha;
      const float p0 = exp2f(t0a[r] - Mn);
      const float p1 = exp2f(t1a[r] - Mn);
      rs[r] = p0 + p1;
      const int rr = r + (hiB ? 8 : 0);
      mysh[rr * 48 + cl]      = (bf16_t)p0;
      mysh[rr * 48 + 16 + cl] = (bf16_t)p1;
    }
#pragma unroll
    for (int off = 1; off < 16; off <<= 1)
#pragma unroll
      for (int r = 0; r < 8; ++r)
        rs[r] += __shfl_xor(rs[r], off, 32);
#pragma unroll
    for (int r = 0; r < 8; ++r) lrun[r] += rs[r];

    // ---- P as A fragment (16x32) from LDS ----
    Frag16 pa;
    {
      const bf16_t* pp = mysh + cl * 48;
      pa.h[0] = *(const v8bf*)(pp + aoff);
      pa.h[1] = *(const v8bf*)(pp + 16 + aoff);
    }
    // ---- O += P @ V over 4 d-tiles (VT slab in LDS) ----
#pragma unroll
    for (int t = 0; t < 4; ++t) {
      const bf16_t* bp = vl + (t * 16 + cl) * 40 + aoff;
      Frag16 b;
      b.h[0] = *(const v8bf*)bp;
      b.h[1] = *(const v8bf*)(bp + 16);
      o[t] = wmma_bf16(pa, b, o[t]);
    }

    __syncthreads();                    // all waves done reading kbuf/vbuf[buf]
    if (w == 0 && kb + 32 < SS) __builtin_amdgcn_s_wait_tensorcnt(0);
    __syncthreads();                    // next slab visible to all waves
    buf ^= 1;
  }

  // ---- normalize and store context bf16 [n, q, h*D + d] ----
#pragma unroll
  for (int r = 0; r < 8; ++r) {
    const float inv = 1.0f / lrun[r];
    const size_t rowoff = ((size_t)n * SS + q0 + r + (hiB ? 8 : 0)) * EE + h * DDIM;
#pragma unroll
    for (int t = 0; t < 4; ++t)
      ctx[rowoff + t * 16 + cl] = (bf16_t)(o[t][r] * inv);
  }
}

// ---------------------------------------------------------------------------
extern "C" void kernel_launch(void* const* d_in, const int* in_sizes, int n_in,
                              void* d_out, int out_size, void* d_ws, size_t ws_size,
                              hipStream_t stream) {
  const float* value = (const float*)d_in[0];
  const float* key_  = (const float*)d_in[1];
  const float* query = (const float*)d_in[2];
  const int*   mask  = (const int*)d_in[3];
  const float* Wv = (const float*)d_in[4];
  const float* bv = (const float*)d_in[5];
  const float* Wk = (const float*)d_in[6];
  const float* bk = (const float*)d_in[7];
  const float* Wq = (const float*)d_in[8];
  const float* bq = (const float*)d_in[9];
  const float* Wo = (const float*)d_in[10];
  const float* bo = (const float*)d_in[11];

  const size_t NSE = (size_t)NB * SS * EE;   // 8,388,608
  const size_t WSZ = (size_t)EE * EE;        // 1,048,576

  char* ws = (char*)d_ws;
  size_t off = 0;
  auto take = [&](size_t bytes) -> char* {
    char* p = ws + off;
    off += (bytes + 255) & ~(size_t)255;
    return p;
  };
  bf16_t* Xv  = (bf16_t*)take(NSE * 2);
  bf16_t* Xk  = (bf16_t*)take(NSE * 2);
  bf16_t* Xq  = (bf16_t*)take(NSE * 2);
  bf16_t* Wvb = (bf16_t*)take(WSZ * 2);
  bf16_t* Wkb = (bf16_t*)take(WSZ * 2);
  bf16_t* Wqb = (bf16_t*)take(WSZ * 2);
  bf16_t* Wob = (bf16_t*)take(WSZ * 2);
  bf16_t* qb   = (bf16_t*)take(NSE * 2);
  bf16_t* kbuf = (bf16_t*)take(NSE * 2);
  bf16_t* vTb  = (bf16_t*)take(NSE * 2);
  bf16_t* ctxb = Xq;   // Xq is dead after the Q projection; reuse its space

  // --- fp32 -> bf16 conversions ---
  const int actBlocks = (int)(NSE / 4 / 256);   // 8192
  const int wBlocks   = (int)(WSZ / 4 / 256);   // 1024
  f32_to_bf16<<<actBlocks, 256, 0, stream>>>(value, Xv, (int)(NSE / 4));
  f32_to_bf16<<<actBlocks, 256, 0, stream>>>(key_,  Xk, (int)(NSE / 4));
  f32_to_bf16<<<actBlocks, 256, 0, stream>>>(query, Xq, (int)(NSE / 4));
  f32_to_bf16<<<wBlocks, 256, 0, stream>>>(Wv, Wvb, (int)(WSZ / 4));
  f32_to_bf16<<<wBlocks, 256, 0, stream>>>(Wk, Wkb, (int)(WSZ / 4));
  f32_to_bf16<<<wBlocks, 256, 0, stream>>>(Wq, Wqb, (int)(WSZ / 4));
  f32_to_bf16<<<wBlocks, 256, 0, stream>>>(Wo, Wob, (int)(WSZ / 4));

  // --- projections: waves = (M/16)*(E/64) = 8192 -> 1024 blocks of 256 ---
  const int M = NB * SS;
  const int gemmBlocks = (M / 16) * (EE / 64) * 32 / 256;   // 1024
  gemm16x64<0><<<gemmBlocks, 256, 0, stream>>>(Xq, Wqb, bq, qb,   M, EE, EE);
  gemm16x64<0><<<gemmBlocks, 256, 0, stream>>>(Xk, Wkb, bk, kbuf, M, EE, EE);
  gemm16x64<1><<<gemmBlocks, 256, 0, stream>>>(Xv, Wvb, bv, vTb,  M, EE, EE);

  // --- attention: waves = N*H*(S/16) = 8192 -> 1024 blocks ---
  attn_fwd<<<1024, 256, 0, stream>>>(qb, kbuf, vTb, mask, ctxb);

  // --- output projection, fp32 result into d_out ---
  gemm16x64<2><<<gemmBlocks, 256, 0, stream>>>(ctxb, Wob, bo, d_out, M, EE, EE);

  (void)in_sizes; (void)n_in; (void)out_size; (void)ws_size;
}
